// LineConv_83708912599664
// MI455X (gfx1250) — matrix operand order repaired and checked by reading, not compiled, not run on hardware
//
#include <hip/hip_runtime.h>
#include <hip/hip_bf16.h>

// ---------------------------------------------------------------------------
// out = s + DA s + (DA)^2 s + (DA)^3 s, computed as 6 skinny GEMMs
//   t = A@x ; x = D@t  (x accumulated into out after each D step)
// BF16 WMMA (v_wmma_f32_16x16x32_bf16), FP32 accumulate. BF16 chosen because
// intermediates reach ~1e17 (FP16 overflows); BF16 keeps FP32 exponent range.
// B operand staged in LDS via global_load_async_to_lds_b128 (ASYNCcnt),
// double-buffered, shared by 4 waves per block. D/A are pre-packed to bf16 in
// the WMMA A-operand layout once per launch (reused 3x each), removing all
// conversion VALU from the GEMM hot loop (ws permitting; fallback converts
// in-register).
// ---------------------------------------------------------------------------

typedef __attribute__((ext_vector_type(16))) __bf16    v16bf;
typedef __attribute__((ext_vector_type(8)))  float     v8f;
typedef __attribute__((ext_vector_type(8)))  unsigned  v8u;

#define BDIM 4096   // B
#define EDIM 128    // EMB
#define LDIM 50     // L
#define KSPL 4      // K-split factor (4096/4 = 1024 per split)
#define WPB  4      // waves (m-tiles) per block sharing the LDS B panel

__device__ __forceinline__ unsigned bf16rne(float f) {
  unsigned u = __builtin_bit_cast(unsigned, f);
  return (u + 0x7FFFu + ((u >> 16) & 1u)) >> 16;   // round-to-nearest-even
}
__device__ __forceinline__ unsigned packbf(float lo, float hi) {
#if __has_builtin(__builtin_amdgcn_cvt_pk_bf16_f32)
  return __builtin_bit_cast(unsigned, __builtin_amdgcn_cvt_pk_bf16_f32(lo, hi));
#else
  return bf16rne(lo) | (bf16rne(hi) << 16);
#endif
}

// 64B async copy global -> LDS (4x b128); INST_OFFSET applies to both sides.
__device__ __forceinline__ void async_copy64(unsigned lds_off, const void* g) {
  unsigned long long ga = (unsigned long long)(size_t)g;
  asm volatile(
      "global_load_async_to_lds_b128 %0, %1, off\n\t"
      "global_load_async_to_lds_b128 %0, %1, off offset:16\n\t"
      "global_load_async_to_lds_b128 %0, %1, off offset:32\n\t"
      "global_load_async_to_lds_b128 %0, %1, off offset:48"
      :: "v"(lds_off), "v"(ga) : "memory");
}
__device__ __forceinline__ void wait_async0() {
  asm volatile("s_wait_asynccnt 0x0" ::: "memory");
}

// 32B B-tile fetch from LDS (2x ds_load_b128)
__device__ __forceinline__ v8u ldtile(const uint4* bt, int t) {
  uint4 b0 = bt[t * 64];
  uint4 b1 = bt[t * 64 + 1];
  return v8u{b0.x, b0.y, b0.z, b0.w, b1.x, b1.y, b1.z, b1.w};
}

// --------------------------- session embedding -----------------------------
__global__ __launch_bounds__(EDIM)
void embed_kernel(const float* __restrict__ emb,      // [100000][128]
                  const int*   __restrict__ si,       // [B][L]
                  const int*   __restrict__ sc,       // [B][L]
                  const float* __restrict__ slen,     // [B][1]
                  float* __restrict__ out,            // [B][128]
                  float* __restrict__ S) {            // [B][128]
  const int b = blockIdx.x;
  const int j = threadIdx.x;
  float sum = 0.f;
  for (int l = 0; l < LDIM; ++l) {
    int idx = si[b * LDIM + l] + sc[b * LDIM + l];    // table row; 0 = zero pad
    if (idx > 0) sum += emb[(long)(idx - 1) * EDIM + j];
  }
  float v = sum / slen[b];
  out[b * EDIM + j] = v;
  S[b * EDIM + j]   = v;
}

// ------------------------------ X packing ----------------------------------
// bf16, pre-swizzled into the WMMA 32x16 B-operand layout:
// dword[((p*8+t)*32+lane)*8+v] = {K, K+1} with K = 32p + 16*(v/4) + 8*(lane/16) + 2*(v%4),
// column n = 16t + lane%16. Sums `nparts` fp32 partial slices while packing.
__global__ __launch_bounds__(256)
void pack_kernel(const float* __restrict__ X,         // [nparts][4096][128]
                 int nparts,
                 unsigned* __restrict__ Xpk) {        // 262144 dwords
  const int tid = blockIdx.x * 256 + threadIdx.x;     // < 262144
  const int v = tid & 7;
  const int L = (tid >> 3) & 31;
  const int t = (tid >> 8) & 7;
  const int p = tid >> 11;                            // 0..127
  const int n = t * 16 + (L & 15);
  const int K = p * 32 + ((v >> 2) << 4) + ((L >> 4) << 3) + ((v & 3) << 1);
  float x0 = 0.f, x1 = 0.f;
  for (int q = 0; q < nparts; ++q) {
    const float* src = X + (long)q * (BDIM * EDIM);
    x0 += src[(long)K * EDIM + n];
    x1 += src[(long)(K + 1) * EDIM + n];
  }
  Xpk[tid] = packbf(x0, x1);
}

// ------------------------------ M packing ----------------------------------
// Pre-pack a 4096x4096 fp32 matrix into bf16 in the WMMA 16-bit A-operand
// layout: dword[((r*128+p)*32+L)*8+v] = {K,K+1} of row r*16+(L&15),
// K = 32p + 16*(v/4) + 8*(L/16) + 2*(v%4).
__global__ __launch_bounds__(256)
void packm_kernel(const float* __restrict__ M, unsigned* __restrict__ Mpk) {
  const long tid = (long)blockIdx.x * 256 + threadIdx.x;   // < 8388608
  const int v = (int)tid & 7;
  const int L = ((int)tid >> 3) & 31;
  const int p = ((int)tid >> 8) & 127;
  const int r = (int)(tid >> 15);                          // 0..255
  const int row = r * 16 + (L & 15);
  const int K = p * 32 + ((v >> 2) << 4) + ((L >> 4) << 3) + ((v & 3) << 1);
  const float* src = M + (long)row * BDIM + K;
  Mpk[tid] = packbf(src[0], src[1]);
}

// ------------------------- GEMM (prepacked A) ------------------------------
__global__ __launch_bounds__(32 * WPB)
void gemm_pk_kernel(const unsigned* __restrict__ Mpk,  // bf16 A-operand layout
                    const unsigned* __restrict__ Xpk,  // packed bf16 B operand
                    float*          __restrict__ Y4) { // [KSPL][4096][128]
  __shared__ unsigned shb[2 * 2048];                   // 2 x 8KB panels

  const int tid  = threadIdx.x;
  const int lane = tid & 31;
  const int wv   = tid >> 5;
  const int mt   = blockIdx.x * WPB + wv;              // 0..255
  const int ks   = blockIdx.y;
  const int m    = lane & 15;
  const int lh   = lane >> 4;

  const int NP = 128 / KSPL;
  const int p0 = ks * NP;

  // A stream: 2 contiguous b128 per panel, advance 256 dwords/panel
  const uint4* apk = (const uint4*)(Mpk + (((long)mt * 128 + p0) * 32 + lane) * 8);

  const unsigned  shbase = (unsigned)(size_t)(&shb[0]);
  const unsigned  lds_my = shbase + tid * 64;
  const unsigned* gsrc   = Xpk + p0 * 2048 + tid * 16;
  const unsigned* shl    = &shb[lane * 8];

  v8f acc[8];
  #pragma unroll
  for (int t = 0; t < 8; ++t) acc[t] = v8f{};

  async_copy64(lds_my, gsrc);
  wait_async0();
  __syncthreads();

  for (int pp = 0; pp < NP; ++pp) {
    const int buf = pp & 1;
    if (pp + 1 < NP)
      async_copy64(lds_my + ((buf ^ 1) << 13), gsrc + (pp + 1) * 2048);

    uint4 a0 = apk[0], a1 = apk[1];                    // 2x global_load_b128
    v8u au = {a0.x, a0.y, a0.z, a0.w, a1.x, a1.y, a1.z, a1.w};
    const v16bf a = __builtin_bit_cast(v16bf, au);
    apk += 64;

    // distance-1 pipeline: load tile t+1 before WMMA t (2 live B tiles)
    const uint4* bt = (const uint4*)(shl + (buf << 11));
    v8u cur = ldtile(bt, 0);
    #pragma unroll
    for (int t = 0; t < 8; ++t) {
      v8u nxt = cur;
      if (t < 7) nxt = ldtile(bt, t + 1);
      acc[t] = __builtin_amdgcn_wmma_f32_16x16x32_bf16(
          false, a, false, __builtin_bit_cast(v16bf, cur),
          (short)0, acc[t], false, false);
      cur = nxt;
    }

    wait_async0();
    __syncthreads();
  }

  float* ybase = Y4 + (long)ks * (BDIM * EDIM);
  const int row0 = mt * 16 + (lh << 3);
  #pragma unroll
  for (int t = 0; t < 8; ++t) {
    const int col = t * 16 + m;
    #pragma unroll
    for (int g = 0; g < 8; ++g)
      ybase[(long)(row0 + g) * EDIM + col] = acc[t][g];
  }
}

// --------------------- GEMM (convert-on-the-fly fallback) ------------------
__global__ __launch_bounds__(32 * WPB)
void gemm_kernel(const float*    __restrict__ M,      // [4096][4096] row-major
                 const unsigned* __restrict__ Xpk,
                 float*          __restrict__ Y4) {
  __shared__ unsigned shb[2 * 2048];

  const int tid  = threadIdx.x;
  const int lane = tid & 31;
  const int wv   = tid >> 5;
  const int mt   = blockIdx.x * WPB + wv;
  const int ks   = blockIdx.y;
  const int m    = lane & 15;
  const int lh   = lane >> 4;

  const int NP = 128 / KSPL;
  const int p0 = ks * NP;

  const float* ap = M + (long)(mt * 16 + m) * BDIM + p0 * 32 + (lh << 3);

  const unsigned  shbase = (unsigned)(size_t)(&shb[0]);
  const unsigned  lds_my = shbase + tid * 64;
  const unsigned* gsrc   = Xpk + p0 * 2048 + tid * 16;
  const unsigned* shl    = &shb[lane * 8];

  v8f acc[8];
  #pragma unroll
  for (int t = 0; t < 8; ++t) acc[t] = v8f{};

  async_copy64(lds_my, gsrc);
  wait_async0();
  __syncthreads();

  for (int pp = 0; pp < NP; ++pp) {
    const int buf = pp & 1;
    if (pp + 1 < NP)
      async_copy64(lds_my + ((buf ^ 1) << 13), gsrc + (pp + 1) * 2048);

    v8u au;
    #pragma unroll
    for (int v = 0; v < 8; ++v) {
      const int c0 = ((v >> 2) << 4) + ((v & 3) << 1); // {0,2,4,6,16,18,20,22}
      au[v] = packbf(ap[c0], ap[c0 + 1]);
    }
    const v16bf a = __builtin_bit_cast(v16bf, au);
    ap += 32;

    const uint4* bt = (const uint4*)(shl + (buf << 11));
    v8u cur = ldtile(bt, 0);
    #pragma unroll
    for (int t = 0; t < 8; ++t) {
      v8u nxt = cur;
      if (t < 7) nxt = ldtile(bt, t + 1);
      acc[t] = __builtin_amdgcn_wmma_f32_16x16x32_bf16(
          false, a, false, __builtin_bit_cast(v16bf, cur),
          (short)0, acc[t], false, false);
      cur = nxt;
    }

    wait_async0();
    __syncthreads();
  }

  float* ybase = Y4 + (long)ks * (BDIM * EDIM);
  const int row0 = mt * 16 + (lh << 3);
  #pragma unroll
  for (int t = 0; t < 8; ++t) {
    const int col = t * 16 + m;
    #pragma unroll
    for (int g = 0; g < 8; ++g)
      ybase[(long)(row0 + g) * EDIM + col] = acc[t][g];
  }
}

// --------------------------- out += sum(partials) --------------------------
__global__ __launch_bounds__(256)
void add_kernel(const float* __restrict__ Y4, float* __restrict__ out) {
  const int e = blockIdx.x * 256 + threadIdx.x;       // < 524288
  float s = 0.f;
  #pragma unroll
  for (int q = 0; q < KSPL; ++q) s += Y4[e + q * (BDIM * EDIM)];
  out[e] += s;
}

// ---------------------------------------------------------------------------
extern "C" void kernel_launch(void* const* d_in, const int* in_sizes, int n_in,
                              void* d_out, int out_size, void* d_ws, size_t ws_size,
                              hipStream_t stream) {
  const float* emb  = (const float*)d_in[0];
  const float* D    = (const float*)d_in[1];
  const float* A    = (const float*)d_in[2];
  const int*   si   = (const int*)d_in[3];
  const int*   sc   = (const int*)d_in[4];
  const float* slen = (const float*)d_in[5];
  float* out = (float*)d_out;

  char* ws = (char*)d_ws;
  float*    S   = (float*)ws;                          //  2 MB
  float*    Y4a = (float*)(ws + (2u  << 20));          //  8 MB [KSPL][4096][128]
  float*    Y4b = (float*)(ws + (10u << 20));          //  8 MB
  unsigned* Xpk = (unsigned*)(ws + (18u << 20));       //  1 MB
  unsigned* Dpk = (unsigned*)(ws + (20u << 20));       // 32 MB (optional)
  unsigned* Apk = (unsigned*)(ws + (52u << 20));       // 32 MB (optional)
  const bool prepk = ws_size >= ((size_t)84 << 20);

  embed_kernel<<<BDIM, EDIM, 0, stream>>>(emb, si, sc, slen, out, S);

  if (prepk) {                                         // once; reused 3x each
    packm_kernel<<<32768, 256, 0, stream>>>(D, Dpk);
    packm_kernel<<<32768, 256, 0, stream>>>(A, Apk);
  }

  const float*    Mf[6] = {A, D, A, D, A, D};
  const unsigned* Mp[6] = {Apk, Dpk, Apk, Dpk, Apk, Dpk};
  float* ybufs[2] = {Y4a, Y4b};
  const float* src = S;
  int parts = 1;
  dim3 ggrid(BDIM / 16 / WPB, KSPL);                   // 64 x 4 blocks

  for (int s6 = 0; s6 < 6; ++s6) {                     // layers = 3 -> 6 GEMMs
    pack_kernel<<<262144 / 256, 256, 0, stream>>>(src, parts, Xpk);
    float* Y = ybufs[s6 & 1];
    if (prepk)
      gemm_pk_kernel<<<ggrid, 32 * WPB, 0, stream>>>(Mp[s6], Xpk, Y);
    else
      gemm_kernel<<<ggrid, 32 * WPB, 0, stream>>>(Mf[s6], Xpk, Y);
    if (s6 & 1) add_kernel<<<(BDIM * EDIM) / 256, 256, 0, stream>>>(Y, out);
    src = Y;
    parts = KSPL;
  }
}
